// CrossAttention_63625645522957
// MI455X (gfx1250) — matrix operand order repaired and checked by reading, not compiled, not run on hardware
//
#include <hip/hip_runtime.h>

// ---------------------------------------------------------------- constants
#define B_   4
#define LQ_  2048
#define LK_  2048
#define E_   768
#define H_   12
#define D_   64
#define SCALE_ 0.03125f            // 0.25 / sqrt(64)
#define LN_EPS_ 1e-5f

typedef _Float16 h16;
typedef _Float16 v8h  __attribute__((ext_vector_type(8)));
typedef _Float16 v16h __attribute__((ext_vector_type(16)));
typedef float    v8f  __attribute__((ext_vector_type(8)));
typedef int      v4i  __attribute__((ext_vector_type(4)));

typedef __attribute__((address_space(1))) v4i AS1_v4i;   // global
typedef __attribute__((address_space(3))) v4i AS3_v4i;   // LDS

__device__ inline v16h ld16(const h16* lo, const h16* hi) {
    v8h a = *(const v8h*)lo;
    v8h b = *(const v8h*)hi;
    return __builtin_shufflevector(a, b, 0,1,2,3,4,5,6,7,8,9,10,11,12,13,14,15);
}

__device__ inline v8f wmma_f16(v16h a, v16h b, v8f c) {
    return __builtin_amdgcn_wmma_f32_16x16x32_f16(
        /*neg_a=*/false, a, /*neg_b=*/false, b,
        /*c_mod=*/(short)0, c, /*reuse_a=*/false, /*reuse_b=*/false);
}

// ------------------------------------------------ async global->LDS copy (16B per lane)
// gfx1250 GLOBAL_LOAD_ASYNC_TO_LDS_B128, tracked by ASYNCcnt. Guarded so the file
// still compiles (sync fallback) if this toolchain doesn't declare the builtins.
#if defined(__HIP_DEVICE_COMPILE__) && \
    __has_builtin(__builtin_amdgcn_global_load_async_to_lds_b128) && \
    __has_builtin(__builtin_amdgcn_s_wait_asynccnt)
#define USE_ASYNC_LDS 1
#else
#define USE_ASYNC_LDS 0
#endif

__device__ inline void cp16(const h16* g, h16* l) {
#if USE_ASYNC_LDS
    v4i* gv = (v4i*)g;                  // strip const, reinterpret as 16B vector
    v4i* lv = (v4i*)l;
    __builtin_amdgcn_global_load_async_to_lds_b128(
        (AS1_v4i*)gv, (AS3_v4i*)lv, 0, 0);
#else
    *(v8h*)l = *(const v8h*)g;
#endif
}

__device__ inline void cp_wait() {
#if USE_ASYNC_LDS
    __builtin_amdgcn_s_wait_asynccnt(0);
#endif
}

// ---------------------------------------------------------------- LayerNorm + cast to f16
__global__ void __launch_bounds__(256)
ln_cast_kernel(const float* __restrict__ x, const float* __restrict__ gam,
               const float* __restrict__ bet, h16* __restrict__ out) {
    __shared__ float s1[256], s2[256];
    const int row = blockIdx.x, tid = threadIdx.x;
    const float* xr = x + (size_t)row * E_;
    float v0 = xr[tid], v1 = xr[tid + 256], v2 = xr[tid + 512];
    s1[tid] = v0 + v1 + v2;
    s2[tid] = v0 * v0 + v1 * v1 + v2 * v2;
    __syncthreads();
    for (int off = 128; off > 0; off >>= 1) {
        if (tid < off) { s1[tid] += s1[tid + off]; s2[tid] += s2[tid + off]; }
        __syncthreads();
    }
    const float mu  = s1[0] * (1.f / E_);
    const float var = s2[0] * (1.f / E_) - mu * mu;
    const float rs  = rsqrtf(var + LN_EPS_);
    h16* orow = out + (size_t)row * E_;
    orow[tid]       = (h16)((v0 - mu) * rs * gam[tid]       + bet[tid]);
    orow[tid + 256] = (h16)((v1 - mu) * rs * gam[tid + 256] + bet[tid + 256]);
    orow[tid + 512] = (h16)((v2 - mu) * rs * gam[tid + 512] + bet[tid + 512]);
}

// ---------------------------------------------------------------- f32 -> f16 cast
__global__ void cast_f2h_kernel(const float* __restrict__ in, h16* __restrict__ out, int n) {
    int i = blockIdx.x * blockDim.x + threadIdx.x;
    if (i < n) out[i] = (h16)in[i];
}

// ---------------------------------------------------------------- WMMA GEMM  C = A(MxK,f16) @ B(KxN,f16)
// Block: 256 thr = 8 waves -> 64x64 tile. Wave (wm 0..3, wn 0..1) -> 16x32, 2 accumulators.
template <bool FINAL>
__global__ void __launch_bounds__(256)
gemm_wmma_kernel(const h16* __restrict__ A, const h16* __restrict__ Bm,
                 h16* __restrict__ Ch, float* __restrict__ Cf,
                 const float* __restrict__ bias, const float* __restrict__ gainp,
                 int M, int N, int K) {
    __shared__ __align__(16) h16 As[64][40];   // [m][k]  (80B rows, 16B multiple)
    __shared__ __align__(16) h16 Bs[64][40];   // [n][k]  transposed for K-contiguous frags
    const int tid  = threadIdx.x;
    const int lane = tid & 31, w = tid >> 5;
    const int wm = w & 3, wn = w >> 2;
    const int g = lane >> 4, ln = lane & 15;
    const int m0 = blockIdx.y * 64, n0 = blockIdx.x * 64;

    v8f acc0 = {}; v8f acc1 = {};

    for (int k0 = 0; k0 < K; k0 += 32) {
        __syncthreads();
        {   // A tile: 64x32 halves -> async 16B per thread
            int row = tid >> 2, off = (tid & 3) * 8;
            cp16(&A[(size_t)(m0 + row) * K + k0 + off], &As[row][off]);
        }
        {   // B tile transposed: each thread gathers 8 k-values of one column
            int n = tid & 63, kc = tid >> 6;           // kc 0..3
            v8h t;
            #pragma unroll
            for (int j = 0; j < 8; ++j)
                t[j] = Bm[(size_t)(k0 + kc * 8 + j) * N + n0 + n];
            *(v8h*)&Bs[n][kc * 8] = t;
        }
        cp_wait();
        __syncthreads();

        // A frag: k = i + 16g (contiguous)
        const h16* ap = &As[16 * wm + ln][0];
        v16h a = ld16(ap + 16 * g, ap + 16 * g + 8);
        // B frags: k = i + 16g (contiguous)
        const h16* bp0 = &Bs[32 * wn + ln][0];
        const h16* bp1 = &Bs[32 * wn + 16 + ln][0];
        v16h b0 = ld16(bp0 + 16 * g, bp0 + 16 * g + 8);
        v16h b1 = ld16(bp1 + 16 * g, bp1 + 16 * g + 8);
        acc0 = wmma_f16(a, b0, acc0);
        acc1 = wmma_f16(a, b1, acc1);
    }

    const float gn = FINAL ? gainp[0] : 1.f;
    #pragma unroll
    for (int cb = 0; cb < 2; ++cb) {
        v8f acc = cb ? acc1 : acc0;
        #pragma unroll
        for (int r = 0; r < 8; ++r) {
            int row = m0 + 16 * wm + r + 8 * g;
            int col = n0 + 32 * wn + cb * 16 + ln;
            if (FINAL) Cf[(size_t)row * N + col] = gn * (acc[r] + bias[col]);
            else       Ch[(size_t)row * N + col] = (h16)acc[r];
        }
    }
}

// ---------------------------------------------------------------- RoPE + [B,L,H*D] -> [B,H,L,D]
__global__ void rope_reshape_kernel(const h16* __restrict__ src, const float* __restrict__ fr,
                                    h16* __restrict__ dst, int srcStride, int colOff) {
    int idx = blockIdx.x * blockDim.x + threadIdx.x;
    const int total = B_ * H_ * LQ_ * (D_ / 2);
    if (idx >= total) return;
    int d2 = idx & 31;
    int l  = (idx >> 5) & (LQ_ - 1);
    int t  = idx >> 16;
    int h  = t % H_;
    int b  = t / H_;
    size_t si = (size_t)(b * LQ_ + l) * srcStride + h * D_ + 2 * d2 + colOff;
    float x0 = (float)src[si], x1 = (float)src[si + 1];
    const float* f = fr + ((size_t)l * 32 + d2) * 2;
    float c = f[0], s = f[1];
    float r0 = x0 * c - x1 * s;
    float r1 = x0 * s + x1 * c;
    size_t di = ((size_t)(b * H_ + h) * LQ_ + l) * D_ + 2 * d2;
    dst[di]     = (h16)r0;
    dst[di + 1] = (h16)r1;
}

// ---------------------------------------------------------------- V -> V^T  [B,H,D,LK] (coalesced both sides)
__global__ void __launch_bounds__(256)
v_transpose_kernel(const h16* __restrict__ kv, h16* __restrict__ vt) {
    __shared__ __align__(16) h16 t[64][72];
    const int tid = threadIdx.x;
    const int lt = blockIdx.x * 64, bh = blockIdx.y;
    const int b = bh / H_, h = bh % H_;
    #pragma unroll
    for (int j = 0; j < 2; ++j) {
        int cidx = tid + j * 256;                  // 0..511 chunks of 8 halves
        int lr = cidx >> 3, off = (cidx & 7) * 8;
        *(v8h*)&t[lr][off] =
            *(const v8h*)&kv[(size_t)(b * LQ_ + lt + lr) * (2 * E_) + E_ + h * D_ + off];
    }
    __syncthreads();
    #pragma unroll
    for (int j = 0; j < 2; ++j) {
        int cidx = tid + j * 256;
        int dr = cidx >> 3, off = (cidx & 7) * 8;
        v8h o;
        #pragma unroll
        for (int u = 0; u < 8; ++u) o[u] = t[off + u][dr];
        *(v8h*)&vt[((size_t)bh * D_ + dr) * LK_ + lt + off] = o;
    }
}

// ---------------------------------------------------------------- Flash attention (double-buffered async KV)
// Block: 128 thr = 4 waves; block tile = 64 q rows; each wave owns 16 q rows x all 64 d.
// KV loop: 64-wide tiles; tile t+1 streams global->LDS (ASYNCcnt) while tile t computes.
__global__ void __launch_bounds__(128)
attn_kernel(const h16* __restrict__ Q, const h16* __restrict__ K,
            const h16* __restrict__ Vt, h16* __restrict__ out) {
    __shared__ __align__(16) h16 Kd[2][64][72];   // [buf][k_col][d]
    __shared__ __align__(16) h16 Vd[2][64][72];   // [buf][d][k_row]   (V pre-transposed in global)
    __shared__ __align__(16) h16 Ps[4][16][72];   // per-wave P tile

    const int tid  = threadIdx.x;
    const int lane = tid & 31, w = tid >> 5;
    const int g = lane >> 4, ln = lane & 15;
    const int qtile = blockIdx.x, bh = blockIdx.y;
    const int b = bh / H_, h = bh % H_;
    const int qbase = qtile * 64 + 16 * w;

    // Q fragments for this wave's 16 rows (registers for the whole KV loop)
    const h16* qrow = Q + ((size_t)bh * LQ_ + qbase + ln) * D_;
    v16h aq0 = ld16(qrow + 16 * g,      qrow + 16 * g + 8);
    v16h aq1 = ld16(qrow + 32 + 16 * g, qrow + 32 + 16 * g + 8);

    float m[8], l[8];
    v8f o[4];
    #pragma unroll
    for (int r = 0; r < 8; ++r) { m[r] = -1e30f; l[r] = 0.f; }
    #pragma unroll
    for (int cb = 0; cb < 4; ++cb) o[cb] = (v8f){};

    auto issue = [&](int kt, int buf) {
        #pragma unroll
        for (int j = 0; j < 4; ++j) {             // 512 chunks of 8 halves per tile pair
            int cidx = tid + j * 128;
            int r = cidx >> 3, off = (cidx & 7) * 8;
            cp16(&K [((size_t)bh * LK_ + kt + r) * D_  + off], &Kd[buf][r][off]);
            cp16(&Vt[((size_t)bh * D_  + r) * LK_ + kt + off], &Vd[buf][r][off]);
        }
    };

    issue(0, 0);
    int buf = 0;
    for (int kt = 0; kt < LK_; kt += 64, buf ^= 1) {
        cp_wait();                                 // my async copies for Kd/Vd[buf] done
        __syncthreads();                           // everyone's copies done / prev reads done
        if (kt + 64 < LK_) issue(kt + 64, buf ^ 1);

        // S = Q @ K^T  (k-dim = d = 64 -> two x32 steps per 16-col block)
        v8f s[4];
        #pragma unroll
        for (int cb = 0; cb < 4; ++cb) {
            const h16* bp = &Kd[buf][cb * 16 + ln][0];
            v16h b0 = ld16(bp + 16 * g,      bp + 16 * g + 8);
            v16h b1 = ld16(bp + 32 + 16 * g, bp + 32 + 16 * g + 8);
            v8f c = {};
            c = wmma_f16(aq0, b0, c);
            c = wmma_f16(aq1, b1, c);
            s[cb] = c;
        }

        // online softmax; row r lives on the 16 lanes of this lane's group
        #pragma unroll
        for (int r = 0; r < 8; ++r) {
            float mx = -1e30f;
            #pragma unroll
            for (int cb = 0; cb < 4; ++cb) mx = fmaxf(mx, s[cb][r]);
            mx *= SCALE_;
            for (int off = 1; off < 16; off <<= 1) mx = fmaxf(mx, __shfl_xor(mx, off, 32));
            float mnew = fmaxf(m[r], mx);
            float corr = __expf(m[r] - mnew);
            float psum = 0.f;
            #pragma unroll
            for (int cb = 0; cb < 4; ++cb) {
                float p = __expf(s[cb][r] * SCALE_ - mnew);
                Ps[w][r + 8 * g][cb * 16 + ln] = (h16)p;
                psum += p;
            }
            for (int off = 1; off < 16; off <<= 1) psum += __shfl_xor(psum, off, 32);
            l[r] = l[r] * corr + psum;
            m[r] = mnew;
            #pragma unroll
            for (int cb = 0; cb < 4; ++cb) o[cb][r] *= corr;
        }

        // O += P @ V   (own wave's Ps region; per-wave LDS ops are in-order)
        const h16* pp = &Ps[w][ln][0];
        v16h ap0 = ld16(pp + 16 * g,      pp + 16 * g + 8);
        v16h ap1 = ld16(pp + 32 + 16 * g, pp + 32 + 16 * g + 8);
        #pragma unroll
        for (int cb = 0; cb < 4; ++cb) {
            const h16* vp = &Vd[buf][cb * 16 + ln][0];
            v16h b0 = ld16(vp + 16 * g,      vp + 16 * g + 8);
            v16h b1 = ld16(vp + 32 + 16 * g, vp + 32 + 16 * g + 8);
            o[cb] = wmma_f16(ap0, b0, o[cb]);
            o[cb] = wmma_f16(ap1, b1, o[cb]);
        }
    }

    // normalize, write back to [B, LQ, H*D] layout for the output projection
    #pragma unroll
    for (int cb = 0; cb < 4; ++cb)
        #pragma unroll
        for (int r = 0; r < 8; ++r) {
            int qr = qbase + r + 8 * g;
            out[((size_t)b * LQ_ + qr) * E_ + h * D_ + cb * 16 + ln] = (h16)(o[cb][r] / l[r]);
        }
}

// ---------------------------------------------------------------- launch
extern "C" void kernel_launch(void* const* d_in, const int* in_sizes, int n_in,
                              void* d_out, int out_size, void* d_ws, size_t ws_size,
                              hipStream_t stream) {
    (void)in_sizes; (void)n_in; (void)out_size; (void)ws_size;
    const float* x_q   = (const float*)d_in[0];
    const float* x_kv  = (const float*)d_in[1];
    const float* fq    = (const float*)d_in[2];
    const float* fk    = (const float*)d_in[3];
    const float* Wq    = (const float*)d_in[4];
    const float* Wkv   = (const float*)d_in[5];
    const float* Wo    = (const float*)d_in[6];
    const float* bo    = (const float*)d_in[7];
    const float* lnqg  = (const float*)d_in[8];
    const float* lnqb  = (const float*)d_in[9];
    const float* lnkg  = (const float*)d_in[10];
    const float* lnkb  = (const float*)d_in[11];
    const float* gain  = (const float*)d_in[12];
    float* outp = (float*)d_out;

    const size_t ROWS = (size_t)B_ * LQ_;         // 8192
    h16* p = (h16*)d_ws;
    h16* xq_ln   = p; p += ROWS * E_;
    h16* xkv_ln  = p; p += ROWS * E_;
    h16* wq_h    = p; p += (size_t)E_ * E_;
    h16* wkv_h   = p; p += (size_t)E_ * 2 * E_;
    h16* wo_h    = p; p += (size_t)E_ * E_;
    h16* q_proj  = p; p += ROWS * E_;
    h16* kv_proj = p; p += ROWS * 2 * E_;
    h16* Qh      = p; p += ROWS * E_;
    h16* Kh      = p; p += ROWS * E_;
    h16* VhT     = p; p += ROWS * E_;             // [B,H,D,LK]
    h16* att     = p; p += ROWS * E_;

    // 1) LayerNorm + f16 cast
    ln_cast_kernel<<<dim3((unsigned)ROWS), 256, 0, stream>>>(x_q,  lnqg, lnqb, xq_ln);
    ln_cast_kernel<<<dim3((unsigned)ROWS), 256, 0, stream>>>(x_kv, lnkg, lnkb, xkv_ln);

    // 2) weight casts
    cast_f2h_kernel<<<(E_*E_ + 255)/256, 256, 0, stream>>>(Wq,  wq_h,  E_*E_);
    cast_f2h_kernel<<<(E_*2*E_ + 255)/256, 256, 0, stream>>>(Wkv, wkv_h, E_*2*E_);
    cast_f2h_kernel<<<(E_*E_ + 255)/256, 256, 0, stream>>>(Wo,  wo_h,  E_*E_);

    // 3) projections (WMMA GEMM)
    gemm_wmma_kernel<false><<<dim3(E_/64,   (unsigned)ROWS/64), 256, 0, stream>>>(
        xq_ln,  wq_h,  q_proj,  nullptr, nullptr, nullptr, (int)ROWS, E_,   E_);
    gemm_wmma_kernel<false><<<dim3(2*E_/64, (unsigned)ROWS/64), 256, 0, stream>>>(
        xkv_ln, wkv_h, kv_proj, nullptr, nullptr, nullptr, (int)ROWS, 2*E_, E_);

    // 4) RoPE + head reshape (Q, K); V: transpose to [B,H,D,LK] for async row loads
    const int rtot = B_ * H_ * LQ_ * (D_ / 2);
    rope_reshape_kernel<<<(rtot + 255)/256, 256, 0, stream>>>(q_proj,  fq, Qh, E_,   0);
    rope_reshape_kernel<<<(rtot + 255)/256, 256, 0, stream>>>(kv_proj, fk, Kh, 2*E_, 0);
    v_transpose_kernel<<<dim3(LK_/64, B_*H_), 256, 0, stream>>>(kv_proj, VhT);

    // 5) flash attention (double-buffered async KV streaming)
    attn_kernel<<<dim3(LQ_/64, B_*H_), 128, 0, stream>>>(Qh, Kh, VhT, att);

    // 6) output projection with fused bias + gain -> f32
    gemm_wmma_kernel<true><<<dim3(E_/64, (unsigned)ROWS/64), 256, 0, stream>>>(
        att, wo_h, nullptr, outp, bo, gain, (int)ROWS, E_, E_);
}
